// STARModel_86955907875135
// MI455X (gfx1250) — matrix operand order repaired and checked by reading, not compiled, not run on hardware
//
#include <hip/hip_runtime.h>
#include <cstdint>

#define BN   16384
#define DINN 1024
#define NDOM 8

typedef __attribute__((ext_vector_type(16))) __bf16 v16bf;
typedef __attribute__((ext_vector_type(8)))  __bf16 v8bf;
typedef __attribute__((ext_vector_type(8)))  float  v8f;

// ---------- helpers ----------
static __device__ __forceinline__ __bf16 f2bf(float f) {
    unsigned u = __builtin_bit_cast(unsigned, f);
    unsigned r = u + 0x7FFFu + ((u >> 16) & 1u);   // round-to-nearest-even
    unsigned short h = (unsigned short)(r >> 16);
    return __builtin_bit_cast(__bf16, h);
}

// Async global->LDS 16-byte copy (CDNA5, tracked by ASYNCcnt).
// VDST VGPR holds the LDS byte address; VADDR pair holds the global address.
static __device__ __forceinline__ void async_g2l_b128(void* lds_ptr, const void* gptr) {
    uint32_t lofs = (uint32_t)(uintptr_t)lds_ptr;   // low 32 bits of generic addr == LDS offset
    asm volatile("global_load_async_to_lds_b128 %0, %1, off"
                 :: "v"(lofs), "v"(gptr) : "memory");
}
static __device__ __forceinline__ void wait_async0() {
    asm volatile("s_wait_asynccnt 0" ::: "memory");
}

// A fragment (16x32 bf16, row-major, leading dim lda elements).
// lanes 0-15: row r, K {k0..k0+15}; lanes 16-31: row r, K {k0+16..k0+31}
static __device__ __forceinline__ v16bf load_a_frag(const __bf16* A, int lda, int lane, int k0) {
    int half = lane >> 4, r = lane & 15;
    const __bf16* p = A + r * lda + k0 + half * 16;
    v8bf lo = *(const v8bf*)p;
    v8bf hi = *(const v8bf*)(p + 8);
    return __builtin_shufflevector(lo, hi, 0,1,2,3,4,5,6,7,8,9,10,11,12,13,14,15);
}

// B fragment (32x16 bf16). Weights stored transposed Wt[N][K] row-major (ldb=K),
// so lane n holds 16 contiguous K values of column n0+n (one 32-byte load).
static __device__ __forceinline__ v16bf load_b_frag(const __bf16* Wt, int ldb, int lane, int n0, int k0) {
    int half = lane >> 4, n = lane & 15;
    return *(const v16bf*)(Wt + (size_t)(n0 + n) * ldb + k0 + half * 16);
}

// NT adjacent 16-wide N tiles per wave: one A load shared by NT WMMAs per k-step.
template <int NT>
static __device__ __forceinline__ void tile_gemm_multi(const __bf16* A, int lda,
                                                       const __bf16* Wt, int ldb,
                                                       int n0, int K, int lane, v8f acc[NT]) {
    for (int k0 = 0; k0 < K; k0 += 32) {
        v16bf a = load_a_frag(A, lda, lane, k0);
#pragma unroll
        for (int t = 0; t < NT; ++t) {
            v16bf b = load_b_frag(Wt, ldb, lane, n0 + t * 16, k0);
            acc[t] = __builtin_amdgcn_wmma_f32_16x16x32_bf16(false, a, false, b,
                                                             (short)0, acc[t], false, false);
        }
    }
}

// ---------- prep kernels ----------
__global__ void zero_cnt_kernel(int* cnt) {
    if (threadIdx.x < NDOM) cnt[threadIdx.x] = 0;
}

// (batch,K,N) fp32 -> (batch,N,K) bf16
__global__ void convert_wt_kernel(const float* __restrict__ W, __bf16* __restrict__ Wt, int K, int N) {
    int b = blockIdx.y;
    int idx = blockIdx.x * 256 + threadIdx.x;
    if (idx >= K * N) return;
    size_t sz = (size_t)K * N;
    int k = idx / N, n = idx % N;
    Wt[b * sz + (size_t)n * K + k] = f2bf(W[b * sz + idx]);
}

// aux logit depends only on domain: precompute 8 scalars
__global__ void aux_kernel(const float* __restrict__ emb, const float* __restrict__ Wa1,
                           const float* __restrict__ ba1, const float* __restrict__ Wa2,
                           const float* __restrict__ ba2, float* __restrict__ auxl) {
    if (threadIdx.x != 0) return;
    for (int d = 0; d < NDOM; ++d) {
        float l = ba2[0];
        for (int j = 0; j < 32; ++j) {
            float a = ba1[j];
            for (int i = 0; i < 8; ++i) a += emb[d * 8 + i] * Wa1[i * 32 + j];
            l += fmaxf(a, 0.f) * Wa2[j];
        }
        auxl[d] = l;
    }
}

// ---------- layernorm + domain affine + bf16 + bucketize ----------
__global__ __launch_bounds__(256) void ln_kernel(const float* __restrict__ x, const int* __restrict__ dom,
                          const float* __restrict__ pnw, const float* __restrict__ pnb,
                          __bf16* __restrict__ xn, int* __restrict__ cnt, int* __restrict__ perm) {
    __shared__ float rs[256], rq[256];
    int s = blockIdx.x, tid = threadIdx.x;
    float4 v = ((const float4*)(x + (size_t)s * DINN))[tid];
    rs[tid] = v.x + v.y + v.z + v.w;
    rq[tid] = v.x * v.x + v.y * v.y + v.z * v.z + v.w * v.w;
    __syncthreads();
    for (int off = 128; off > 0; off >>= 1) {
        if (tid < off) { rs[tid] += rs[tid + off]; rq[tid] += rq[tid + off]; }
        __syncthreads();
    }
    float mean = rs[0] * (1.0f / DINN);
    float var  = rq[0] * (1.0f / DINN) - mean * mean;   // biased, matches jnp.var
    float rstd = rsqrtf(var + 1e-5f);
    int d = dom[s];
    const float* w = pnw + (size_t)d * DINN;
    const float* bb = pnb + (size_t)d * DINN;
    float e[4] = {v.x, v.y, v.z, v.w};
    ushort4 o; unsigned short* op = (unsigned short*)&o;
#pragma unroll
    for (int j = 0; j < 4; ++j) {
        int idx = tid * 4 + j;
        float t = (e[j] - mean) * rstd * w[idx] + bb[idx];
        op[j] = __builtin_bit_cast(unsigned short, f2bf(t));
    }
    *((ushort4*)(xn + (size_t)s * DINN) + tid) = o;
    if (tid == 0) {
        int pos = atomicAdd(cnt + d, 1);
        perm[d * BN + pos] = s;
    }
}

// LDS layout (bytes): xa 16x1032 bf16 @0 (33024) | h1 16x520 bf16 @33024 (16640) | h2 16x264 bf16 @49664 (8448)
#define SMEM_BYTES 58112

// ---------- center chain: 1024 -> 512 -> 256 -> 128 ----------
__global__ __launch_bounds__(256) void center_kernel(
    const __bf16* __restrict__ xn,
    const __bf16* __restrict__ W1t, const float* __restrict__ b1,
    const __bf16* __restrict__ W2t, const float* __restrict__ b2,
    const __bf16* __restrict__ W3t, const float* __restrict__ b3,
    float* __restrict__ hc) {
    __shared__ __align__(16) char smem[SMEM_BYTES];
    __bf16* xa = (__bf16*)smem;
    __bf16* h1 = (__bf16*)(smem + 33024);
    __bf16* h2 = (__bf16*)(smem + 49664);
    int tid = threadIdx.x, wave = tid >> 5, lane = tid & 31;
    int half = lane >> 4, nl = lane & 15;
    int m0 = blockIdx.x * 16;
    for (int c = tid; c < 2048; c += 256) {              // async stage 16x1024 bf16 (32KB)
        int row = c >> 7, pos = c & 127;
        async_g2l_b128(xa + row * 1032 + pos * 8,
                       xn + (size_t)(m0 + row) * DINN + pos * 8);
    }
    wait_async0();
    __syncthreads();
    {                                                    // layer 1: N=512, K=1024 (4 tiles/wave)
        v8f acc[4] = {};
        int n0 = wave * 64;
        tile_gemm_multi<4>(xa, 1032, W1t, 1024, n0, 1024, lane, acc);
#pragma unroll
        for (int t = 0; t < 4; ++t) {
            int n = n0 + t * 16 + nl;
            float bv = b1[n];
#pragma unroll
            for (int i = 0; i < 8; ++i)
                h1[(i + 8 * half) * 520 + n] = f2bf(fmaxf(acc[t][i] + bv, 0.f));
        }
    }
    __syncthreads();
    {                                                    // layer 2: N=256, K=512 (2 tiles/wave)
        v8f acc[2] = {};
        int n0 = wave * 32;
        tile_gemm_multi<2>(h1, 520, W2t, 512, n0, 512, lane, acc);
#pragma unroll
        for (int t = 0; t < 2; ++t) {
            int n = n0 + t * 16 + nl;
            float bv = b2[n];
#pragma unroll
            for (int i = 0; i < 8; ++i)
                h2[(i + 8 * half) * 264 + n] = f2bf(fmaxf(acc[t][i] + bv, 0.f));
        }
    }
    __syncthreads();
    {                                                    // layer 3: N=128, K=256 (no relu)
        v8f acc[1] = {};
        int n0 = wave * 16;
        tile_gemm_multi<1>(h2, 264, W3t, 256, n0, 256, lane, acc);
        float bv = b3[n0 + nl];
#pragma unroll
        for (int i = 0; i < 8; ++i)
            hc[(size_t)(m0 + i + 8 * half) * 128 + n0 + nl] = acc[0][i] + bv;
    }
}

// ---------- domain chain (bucketed) + STAR fusion + head + sigmoid ----------
__global__ __launch_bounds__(256) void domain_kernel(
    const __bf16* __restrict__ xn,
    const __bf16* __restrict__ W1t, const float* __restrict__ b1,   // (ND,512,1024),(ND,512)
    const __bf16* __restrict__ W2t, const float* __restrict__ b2,   // (ND,256,512),(ND,256)
    const __bf16* __restrict__ W3t, const float* __restrict__ b3,   // (ND,128,256),(ND,128)
    const float* __restrict__ hc,
    const float* __restrict__ Wf1, const float* __restrict__ bf1,
    const float* __restrict__ Wf2, const float* __restrict__ bf2,
    const float* __restrict__ auxl,
    const int* __restrict__ cnt, const int* __restrict__ perm,
    float* __restrict__ out) {
    int d = blockIdx.y;
    int nrows = cnt[d];
    int t0 = blockIdx.x * 16;
    if (t0 >= nrows) return;
    __shared__ __align__(16) char smem[SMEM_BYTES];
    __shared__ int smp[16];
    __bf16* xa = (__bf16*)smem;
    __bf16* h1 = (__bf16*)(smem + 33024);
    __bf16* h2 = (__bf16*)(smem + 49664);
    float*  hf = (float*)smem;            // 16x132 f32, reuses dead xa space
    float*  z1 = (float*)(smem + 8448);   // 16x72 f32, also in xa space
    int tid = threadIdx.x, wave = tid >> 5, lane = tid & 31;
    int half = lane >> 4, nl = lane & 15;
    if (tid < 16) {
        int idx = t0 + tid; if (idx > nrows - 1) idx = nrows - 1;   // clamp partial tile
        smp[tid] = perm[d * BN + idx];
    }
    __syncthreads();
    for (int c = tid; c < 2048; c += 256) {              // async gather of 16 bucketed rows
        int row = c >> 7, pos = c & 127;
        async_g2l_b128(xa + row * 1032 + pos * 8,
                       xn + (size_t)smp[row] * DINN + pos * 8);
    }
    wait_async0();
    __syncthreads();
    const __bf16* w1 = W1t + (size_t)d * 512 * 1024; const float* bb1 = b1 + d * 512;
    const __bf16* w2 = W2t + (size_t)d * 256 * 512;  const float* bb2 = b2 + d * 256;
    const __bf16* w3 = W3t + (size_t)d * 128 * 256;  const float* bb3 = b3 + d * 128;
    {
        v8f acc[4] = {};
        int n0 = wave * 64;
        tile_gemm_multi<4>(xa, 1032, w1, 1024, n0, 1024, lane, acc);
#pragma unroll
        for (int t = 0; t < 4; ++t) {
            int n = n0 + t * 16 + nl;
            float bv = bb1[n];
#pragma unroll
            for (int i = 0; i < 8; ++i)
                h1[(i + 8 * half) * 520 + n] = f2bf(fmaxf(acc[t][i] + bv, 0.f));
        }
    }
    __syncthreads();
    {
        v8f acc[2] = {};
        int n0 = wave * 32;
        tile_gemm_multi<2>(h1, 520, w2, 512, n0, 512, lane, acc);
#pragma unroll
        for (int t = 0; t < 2; ++t) {
            int n = n0 + t * 16 + nl;
            float bv = bb2[n];
#pragma unroll
            for (int i = 0; i < 8; ++i)
                h2[(i + 8 * half) * 264 + n] = f2bf(fmaxf(acc[t][i] + bv, 0.f));
        }
    }
    __syncthreads();
    {   // layer 3 + STAR fusion: hf = hc * tanh(hd)
        v8f acc[1] = {};
        int n0 = wave * 16;
        tile_gemm_multi<1>(h2, 264, w3, 256, n0, 256, lane, acc);
        float bv = bb3[n0 + nl];
#pragma unroll
        for (int i = 0; i < 8; ++i) {
            int m = i + 8 * half, n = n0 + nl;
            hf[m * 132 + n] = hc[(size_t)smp[m] * 128 + n] * tanhf(acc[0][i] + bv);
        }
    }
    __syncthreads();
    for (int e = tid; e < 16 * 64; e += 256) {           // head layer: 128 -> 64, relu
        int m = e >> 6, j = e & 63;
        float a = bf1[j];
        for (int k = 0; k < 128; ++k) a += hf[m * 132 + k] * Wf1[k * 64 + j];
        z1[m * 72 + j] = fmaxf(a, 0.f);
    }
    __syncthreads();
    if (tid < 16 && (t0 + tid) < nrows) {                // 64 -> 1, + aux, sigmoid
        int m = tid;
        float l = bf2[0] + auxl[d];
        for (int j = 0; j < 64; ++j) l += z1[m * 72 + j] * Wf2[j];
        out[smp[m]] = 1.f / (1.f + expf(-l));
    }
}

extern "C" void kernel_launch(void* const* d_in, const int* in_sizes, int n_in,
                              void* d_out, int out_size, void* d_ws, size_t ws_size,
                              hipStream_t stream) {
    (void)in_sizes; (void)n_in; (void)out_size; (void)ws_size;
    const float* x   = (const float*)d_in[0];
    const int*   dom = (const int*)d_in[1];
    const float* pnw = (const float*)d_in[2];
    const float* pnb = (const float*)d_in[3];
    const float* Wc1 = (const float*)d_in[4];  const float* bc1 = (const float*)d_in[5];
    const float* Wc2 = (const float*)d_in[6];  const float* bc2 = (const float*)d_in[7];
    const float* Wc3 = (const float*)d_in[8];  const float* bc3 = (const float*)d_in[9];
    const float* Wd1 = (const float*)d_in[10]; const float* bd1 = (const float*)d_in[11];
    const float* Wd2 = (const float*)d_in[12]; const float* bd2 = (const float*)d_in[13];
    const float* Wd3 = (const float*)d_in[14]; const float* bd3 = (const float*)d_in[15];
    const float* Wf1 = (const float*)d_in[16]; const float* bf1 = (const float*)d_in[17];
    const float* Wf2 = (const float*)d_in[18]; const float* bf2 = (const float*)d_in[19];
    const float* emb = (const float*)d_in[20];
    const float* Wa1 = (const float*)d_in[21]; const float* ba1 = (const float*)d_in[22];
    const float* Wa2 = (const float*)d_in[23]; const float* ba2 = (const float*)d_in[24];
    float* out = (float*)d_out;

    char* ws = (char*)d_ws; size_t off = 0;
    auto alloc = [&](size_t bytes) { void* p = ws + off; off += (bytes + 255) & ~(size_t)255; return p; };
    __bf16* xn   = (__bf16*)alloc((size_t)BN * DINN * 2);
    float*  hc   = (float*) alloc((size_t)BN * 128 * 4);
    __bf16* Wc1t = (__bf16*)alloc((size_t)512 * 1024 * 2);
    __bf16* Wc2t = (__bf16*)alloc((size_t)256 * 512 * 2);
    __bf16* Wc3t = (__bf16*)alloc((size_t)128 * 256 * 2);
    __bf16* Wd1t = (__bf16*)alloc((size_t)NDOM * 512 * 1024 * 2);
    __bf16* Wd2t = (__bf16*)alloc((size_t)NDOM * 256 * 512 * 2);
    __bf16* Wd3t = (__bf16*)alloc((size_t)NDOM * 128 * 256 * 2);
    int*    cnt  = (int*)   alloc(NDOM * 4);
    int*    perm = (int*)   alloc((size_t)NDOM * BN * 4);
    float*  auxl = (float*) alloc(NDOM * 4);

    zero_cnt_kernel<<<1, 32, 0, stream>>>(cnt);
    convert_wt_kernel<<<dim3((1024 * 512 + 255) / 256, 1), 256, 0, stream>>>(Wc1, Wc1t, 1024, 512);
    convert_wt_kernel<<<dim3((512 * 256 + 255) / 256, 1), 256, 0, stream>>>(Wc2, Wc2t, 512, 256);
    convert_wt_kernel<<<dim3((256 * 128 + 255) / 256, 1), 256, 0, stream>>>(Wc3, Wc3t, 256, 128);
    convert_wt_kernel<<<dim3((1024 * 512 + 255) / 256, NDOM), 256, 0, stream>>>(Wd1, Wd1t, 1024, 512);
    convert_wt_kernel<<<dim3((512 * 256 + 255) / 256, NDOM), 256, 0, stream>>>(Wd2, Wd2t, 512, 256);
    convert_wt_kernel<<<dim3((256 * 128 + 255) / 256, NDOM), 256, 0, stream>>>(Wd3, Wd3t, 256, 128);
    aux_kernel<<<1, 32, 0, stream>>>(emb, Wa1, ba1, Wa2, ba2, auxl);
    ln_kernel<<<BN, 256, 0, stream>>>(x, dom, pnw, pnb, xn, cnt, perm);
    center_kernel<<<BN / 16, 256, 0, stream>>>(xn, Wc1t, bc1, Wc2t, bc2, Wc3t, bc3, hc);
    domain_kernel<<<dim3(BN / 16, NDOM), 256, 0, stream>>>(xn, Wd1t, bd1, Wd2t, bd2, Wd3t, bd3,
                                                           hc, Wf1, bf1, Wf2, bf2, auxl, cnt, perm, out);
}
